// MultiHeadAttentionWithRoPE_77575699300610
// MI455X (gfx1250) — matrix operand order repaired and checked by reading, not compiled, not run on hardware
//
#include <hip/hip_runtime.h>
#include <hip/hip_bf16.h>
#include <math.h>

// MultiHeadAttentionWithRoPE for MI455X (gfx1250, wave32, WMMA).
// bf16x3 split-precision WMMA (f32 accumulate). LDS is staged FRAGMENT-MAJOR:
// each lane's WMMA operand fragment is a contiguous 32B slot, so operand
// construction is a single aligned v8u load (2x ds_load_b128) with zero
// register shuffling.

#define T_LEN  3520
#define NHEAD  16
#define HDIM   64
#define NPATCH 220

typedef __attribute__((ext_vector_type(16))) __bf16 v16bf;
typedef __attribute__((ext_vector_type(8)))  float  v8f;
typedef __attribute__((ext_vector_type(8)))  unsigned int v8u;

union BPack { unsigned int u; __bf16 b[2]; };

// Split (x0,x1) into packed-bf16 hi dword and lo (residual) dword.
__device__ __forceinline__ void split_pack(float x0, float x1,
                                           unsigned int& hi, unsigned int& lo) {
  __bf16 h0 = (__bf16)x0, h1 = (__bf16)x1;
  __bf16 l0 = (__bf16)(x0 - (float)h0), l1 = (__bf16)(x1 - (float)h1);
  BPack ph, pl;
  ph.b[0] = h0; ph.b[1] = h1;
  pl.b[0] = l0; pl.b[1] = l1;
  hi = ph.u; lo = pl.u;
}

__device__ __forceinline__ v8f wmma_bf16(v16bf a, v16bf b, v8f c) {
  return __builtin_amdgcn_wmma_f32_16x16x32_bf16(false, a, false, b, (short)0, c,
                                                 false, false);
}

__device__ __forceinline__ v16bf frag_load(const unsigned int* slot) {
  return __builtin_bit_cast(v16bf, *(const v8u*)slot);
}

// ---------------------------------------------------------------------------
// Generic C = A(MxK) * B(KxN) [+ bias], fp32 in/out, bf16x3 WMMA internally.
// Block: 128 threads (4 waves). Tile: 64x64 per WG, 16x64 per wave.
// ---------------------------------------------------------------------------
__global__ __launch_bounds__(128) void gemm_bf16x3(
    const float* __restrict__ A, const float* __restrict__ Bm,
    const float* __restrict__ bias, float* __restrict__ C,
    int M, int N, int K)
{
  // Fragment-major: [fragment][lane][j], 32B slot per lane.
  __shared__ __align__(32) unsigned int sAfh[4][32][8], sAfl[4][32][8];
  __shared__ __align__(32) unsigned int sBfh[4][32][8], sBfl[4][32][8];
  const int tid  = threadIdx.x;
  const int lane = tid & 31;
  const int wave = tid >> 5;
  const int m0   = blockIdx.x * 64;
  const int n0   = blockIdx.y * 64;
  const int rm   = lane & 15;
  const int hi16 = (lane >= 16);

  v8f acc[4] = {};

  for (int k0 = 0; k0 < K; k0 += 32) {
    __syncthreads();
    // A: 64 rows x 16 K-pairs (pairs contiguous -> float2 loads), scatter to
    // fragment slots: frag=row>>4, lane=(row&15)+(p>=8?16:0), j=p&7
#pragma unroll
    for (int i = 0; i < 8; ++i) {
      int idx = tid + i * 128;
      int r = idx >> 4, p = idx & 15;
      const float2 v = *(const float2*)&A[(size_t)(m0 + r) * K + k0 + 2 * p];
      unsigned int hi, lo;
      split_pack(v.x, v.y, hi, lo);
      int fl = (r & 15) + ((p >= 8) ? 16 : 0);
      int fj = p & 7;
      sAfh[r >> 4][fl][fj] = hi;
      sAfl[r >> 4][fl][fj] = lo;
    }
    // B: 16 K-pair rows x 64 cols: frag=c>>4, lane=(c&15)+(p>=8?16:0), j=p&7
#pragma unroll
    for (int i = 0; i < 8; ++i) {
      int idx = tid + i * 128;
      int p = idx >> 6, c = idx & 63;
      float x0 = Bm[(size_t)(k0 + 2 * p) * N + n0 + c];
      float x1 = Bm[(size_t)(k0 + 2 * p + 1) * N + n0 + c];
      unsigned int hi, lo;
      split_pack(x0, x1, hi, lo);
      int fl = (c & 15) + ((p >= 8) ? 16 : 0);
      sBfh[c >> 4][fl][p & 7] = hi;
      sBfl[c >> 4][fl][p & 7] = lo;
    }
    __syncthreads();

    v16bf a_hi = frag_load(sAfh[wave][lane]);
    v16bf a_lo = frag_load(sAfl[wave][lane]);
#pragma unroll
    for (int nt = 0; nt < 4; ++nt) {
      v16bf b_hi = frag_load(sBfh[nt][lane]);
      v16bf b_lo = frag_load(sBfl[nt][lane]);
      acc[nt] = wmma_bf16(a_hi, b_hi, acc[nt]);
      acc[nt] = wmma_bf16(a_hi, b_lo, acc[nt]);
      acc[nt] = wmma_bf16(a_lo, b_hi, acc[nt]);
    }
  }

  const int mr = m0 + wave * 16 + (hi16 ? 8 : 0);
#pragma unroll
  for (int nt = 0; nt < 4; ++nt) {
    int col = n0 + nt * 16 + rm;
    float bv = bias ? bias[col] : 0.0f;
#pragma unroll
    for (int r = 0; r < 8; ++r)
      C[(size_t)(mr + r) * N + col] = acc[nt][r] + bv;
  }
}

// ---------------------------------------------------------------------------
// RMS-norm(q,k) + RoPE + scatter to per-head [H][T][D] f32 layouts.
// One wave per (t, h); lane owns the RoPE pair (d0, d0+1).
// ---------------------------------------------------------------------------
__global__ __launch_bounds__(256) void qkv_post(
    const float* __restrict__ qkv, const float* __restrict__ qn_w,
    const float* __restrict__ kn_w, float* __restrict__ qw,
    float* __restrict__ kw, float* __restrict__ vw)
{
  int gw = blockIdx.x * 8 + (threadIdx.x >> 5);
  int lane = threadIdx.x & 31;
  if (gw >= T_LEN * NHEAD) return;
  int t = gw >> 4;
  int h = gw & 15;
  int d0 = lane * 2;
  const float* base = qkv + (size_t)t * (3 * NHEAD * HDIM) + h * HDIM;
  float q0 = base[d0],        q1 = base[d0 + 1];
  float k0 = base[1024 + d0], k1 = base[1024 + d0 + 1];
  float v0 = base[2048 + d0], v1 = base[2048 + d0 + 1];

  float qs = q0 * q0 + q1 * q1;
  float ks = k0 * k0 + k1 * k1;
#pragma unroll
  for (int off = 16; off >= 1; off >>= 1) {
    qs += __shfl_xor(qs, off);
    ks += __shfl_xor(ks, off);
  }
  float qinv = 1.0f / sqrtf(qs * (1.0f / 64.0f) + 1e-6f);
  float kinv = 1.0f / sqrtf(ks * (1.0f / 64.0f) + 1e-6f);
  float nq0 = q0 * qinv * qn_w[d0], nq1 = q1 * qinv * qn_w[d0 + 1];
  float nk0 = k0 * kinv * kn_w[d0], nk1 = k1 * kinv * kn_w[d0 + 1];

  double theta = pow(10000.0, -(double)d0 / 64.0);
  double ang   = (double)t * theta;
  float c = (float)cos(ang);
  float s = (float)sin(ang);

  size_t o = ((size_t)h * T_LEN + t) * HDIM + d0;
  qw[o]     = nq0 * c - nq1 * s;
  qw[o + 1] = nq1 * c + nq0 * s;
  kw[o]     = nk0 * c - nk1 * s;
  kw[o + 1] = nk1 * c + nk0 * s;
  vw[o]     = v0;
  vw[o + 1] = v1;
}

// ---------------------------------------------------------------------------
// Flash attention with frame-causal mask (frame = t / 220).
// Grid: (qblock 0..54, head 0..15). Block: 128 threads = 4 waves.
// Each wave owns 16 queries; WG shares LDS-staged fragment-major K/V blocks.
// ---------------------------------------------------------------------------
__global__ __launch_bounds__(128) void attn_fwd(
    const float* __restrict__ qw, const float* __restrict__ kw,
    const float* __restrict__ vw, float* __restrict__ att)
{
  // K fragments: [d-step s][key-tile c][lane][j]; V fragments: [d-tile][lane][j]
  __shared__ __align__(32) unsigned int sKfh[2][2][32][8], sKfl[2][2][32][8];
  __shared__ __align__(32) unsigned int sVfh[4][32][8],    sVfl[4][32][8];
  __shared__ __align__(8)  float sP[4][16][34];   // per-wave P relayout (f32)

  const int h    = blockIdx.y;
  const int qb   = blockIdx.x;
  const int tid  = threadIdx.x;
  const int lane = tid & 31;
  const int wave = tid >> 5;
  const int q0   = qb * 64 + wave * 16;
  const int rm   = lane & 15;
  const int hi16 = (lane >= 16);
  const int ahalf = hi16 ? 8 : 0;   // A-layout K offset & C-layout row offset

  const float* qh = qw + (size_t)h * T_LEN * HDIM;
  const float* kh = kw + (size_t)h * T_LEN * HDIM;
  const float* vh = vw + (size_t)h * T_LEN * HDIM;

  // Q A-operands for the two K-steps (d=0..31, 32..63), split bf16. Built once.
  v16bf qa_hi[2], qa_lo[2];
  {
    const float* qr = qh + (size_t)(q0 + rm) * HDIM;
#pragma unroll
    for (int s = 0; s < 2; ++s) {
      v8u uh, ul;
#pragma unroll
      for (int j = 0; j < 8; ++j) {
        int kk = s * 32 + 2 * ahalf + (j << 1);
        unsigned int hi, lo;
        split_pack(qr[kk], qr[kk + 1], hi, lo);
        uh[j] = hi; ul[j] = lo;
      }
      qa_hi[s] = __builtin_bit_cast(v16bf, uh);
      qa_lo[s] = __builtin_bit_cast(v16bf, ul);
    }
  }

  const int fq_wave = (q0 + 15) / NPATCH;                     // wave max frame
  const int limit   = ((qb * 64 + 63) / NPATCH + 1) * NPATCH; // WG key limit
  const int nblocks = (limit + 31) >> 5;

  float mrow[8], lrow[8];
  v8f o[4] = {};
#pragma unroll
  for (int r = 0; r < 8; ++r) { mrow[r] = -3.0e38f; lrow[r] = 0.0f; }

  for (int nb = 0; nb < nblocks; ++nb) {
    const int n0 = nb * 32;
    __syncthreads();
    // Stage K: 32 keys x 32 d-pairs (contiguous -> float2), scatter to
    // fragments: s=p>>4, c=key>>4, lane=(key&15)+((p&15)>=8?16:0), j=p&7
#pragma unroll
    for (int i = 0; i < 8; ++i) {
      int idx = tid + i * 128;
      int r = idx >> 5, p = idx & 31;
      const float2 v = *(const float2*)&kh[(size_t)(n0 + r) * HDIM + 2 * p];
      unsigned int hi, lo;
      split_pack(v.x, v.y, hi, lo);
      int fl = (r & 15) + (((p & 15) >= 8) ? 16 : 0);
      sKfh[p >> 4][r >> 4][fl][p & 7] = hi;
      sKfl[p >> 4][r >> 4][fl][p & 7] = lo;
    }
    // Stage V: 16 n-pairs x 64 d: dt=c>>4, lane=(c&15)+(p>=8?16:0), j=p&7
#pragma unroll
    for (int i = 0; i < 8; ++i) {
      int idx = tid + i * 128;
      int p = idx >> 6, c = idx & 63;
      float x0 = vh[(size_t)(n0 + 2 * p) * HDIM + c];
      float x1 = vh[(size_t)(n0 + 2 * p + 1) * HDIM + c];
      unsigned int hi, lo;
      split_pack(x0, x1, hi, lo);
      int fl = (c & 15) + ((p >= 8) ? 16 : 0);
      sVfh[c >> 4][fl][p & 7] = hi;
      sVfl[c >> 4][fl][p & 7] = lo;
    }
    __syncthreads();
    if (nb + 1 < nblocks) {                   // prefetch next block
      __builtin_prefetch(kh + (size_t)(n0 + 32) * HDIM + tid * 16, 0, 1);
      __builtin_prefetch(vh + (size_t)(n0 + 32) * HDIM + tid * 16, 0, 1);
    }
    if (n0 / NPATCH > fq_wave) continue;      // wave-uniform: fully masked

    // S = Q K^T : two 16x16 tiles over keys n0..n0+31
    v8f sc0 = {}, sc1 = {};
#pragma unroll
    for (int s = 0; s < 2; ++s) {
#pragma unroll
      for (int c = 0; c < 2; ++c) {
        v16bf b_hi = frag_load(sKfh[s][c][lane]);
        v16bf b_lo = frag_load(sKfl[s][c][lane]);
        v8f& sacc = c ? sc1 : sc0;
        sacc = wmma_bf16(qa_hi[s], b_hi, sacc);
        sacc = wmma_bf16(qa_hi[s], b_lo, sacc);
        sacc = wmma_bf16(qa_lo[s], b_hi, sacc);
      }
    }

    // scale + frame-causal mask in C layout, block row-max
    float bm[8];
    const int ncol0 = n0 + rm;
#pragma unroll
    for (int r = 0; r < 8; ++r) {
      int fq = (q0 + ahalf + r) / NPATCH;
      float s0 = sc0[r] * 0.125f;             // D^-0.5
      float s1 = sc1[r] * 0.125f;
      if (ncol0 / NPATCH > fq)        s0 = -3.0e38f;
      if ((ncol0 + 16) / NPATCH > fq) s1 = -3.0e38f;
      sc0[r] = s0; sc1[r] = s1;
      bm[r] = fmaxf(s0, s1);
    }
#pragma unroll
    for (int off = 8; off >= 1; off >>= 1)
#pragma unroll
      for (int r = 0; r < 8; ++r)
        bm[r] = fmaxf(bm[r], __shfl_xor(bm[r], off));

    // online softmax update
    float alpha[8], ps[8];
#pragma unroll
    for (int r = 0; r < 8; ++r) {
      float mn = fmaxf(mrow[r], bm[r]);
      alpha[r] = __expf(mrow[r] - mn);
      mrow[r] = mn;
      float p0 = __expf(sc0[r] - mn);
      float p1 = __expf(sc1[r] - mn);
      sc0[r] = p0; sc1[r] = p1;
      ps[r] = p0 + p1;
    }
#pragma unroll
    for (int off = 8; off >= 1; off >>= 1)
#pragma unroll
      for (int r = 0; r < 8; ++r)
        ps[r] += __shfl_xor(ps[r], off);
#pragma unroll
    for (int r = 0; r < 8; ++r) {
      lrow[r] = lrow[r] * alpha[r] + ps[r];
      o[0][r] *= alpha[r]; o[1][r] *= alpha[r];
      o[2][r] *= alpha[r]; o[3][r] *= alpha[r];
    }

    // relayout P: C layout -> LDS -> A layout (within-wave, DScnt-ordered)
#pragma unroll
    for (int r = 0; r < 8; ++r) {
      sP[wave][ahalf + r][rm]      = sc0[r];
      sP[wave][ahalf + r][16 + rm] = sc1[r];
    }
    v8u uph, upl;
#pragma unroll
    for (int j = 0; j < 8; ++j) {
      int kk = 2 * ahalf + (j << 1);
      const float2 pv = *(const float2*)&sP[wave][rm][kk];
      unsigned int hi, lo;
      split_pack(pv.x, pv.y, hi, lo);
      uph[j] = hi; upl[j] = lo;
    }
    v16bf p_hi = __builtin_bit_cast(v16bf, uph);
    v16bf p_lo = __builtin_bit_cast(v16bf, upl);

    // O += P V  (four 16-wide d-tiles)
#pragma unroll
    for (int dt = 0; dt < 4; ++dt) {
      v16bf b_hi = frag_load(sVfh[dt][lane]);
      v16bf b_lo = frag_load(sVfl[dt][lane]);
      o[dt] = wmma_bf16(p_hi, b_hi, o[dt]);
      o[dt] = wmma_bf16(p_hi, b_lo, o[dt]);
      o[dt] = wmma_bf16(p_lo, b_hi, o[dt]);
    }
  }

  // normalize and write to [T][H*D] layout for the projection GEMM
  float inv[8];
#pragma unroll
  for (int r = 0; r < 8; ++r) inv[r] = 1.0f / lrow[r];
  const int mr = q0 + ahalf;
#pragma unroll
  for (int dt = 0; dt < 4; ++dt) {
    int col = h * HDIM + dt * 16 + rm;
#pragma unroll
    for (int r = 0; r < 8; ++r)
      att[(size_t)(mr + r) * (NHEAD * HDIM) + col] = o[dt][r] * inv[r];
  }
}

// ---------------------------------------------------------------------------
extern "C" void kernel_launch(void* const* d_in, const int* in_sizes, int n_in,
                              void* d_out, int out_size, void* d_ws, size_t ws_size,
                              hipStream_t stream) {
  (void)in_sizes; (void)n_in; (void)out_size; (void)ws_size;
  const float* x     = (const float*)d_in[0];
  const float* Wqkv  = (const float*)d_in[1];
  const float* Wproj = (const float*)d_in[2];
  const float* bproj = (const float*)d_in[3];
  const float* qn_w  = (const float*)d_in[4];
  const float* kn_w  = (const float*)d_in[5];
  float* out = (float*)d_out;
  float* ws  = (float*)d_ws;

  float* qkv = ws;                                   // 3520*3072 f32
  float* qw  = ws + (size_t)T_LEN * 3 * NHEAD * HDIM;
  float* kw  = qw + (size_t)T_LEN * NHEAD * HDIM;
  float* vw  = kw + (size_t)T_LEN * NHEAD * HDIM;
  float* att = ws;                                   // reuse qkv region

  // 1) qkv = x @ Wqkv              (3520x1024 @ 1024x3072)
  gemm_bf16x3<<<dim3(55, 48), 128, 0, stream>>>(x, Wqkv, nullptr, qkv,
                                                T_LEN, 3 * NHEAD * HDIM, NHEAD * HDIM);
  // 2) RMS-norm + RoPE + per-head scatter
  qkv_post<<<(T_LEN * NHEAD) / 8, 256, 0, stream>>>(qkv, qn_w, kn_w, qw, kw, vw);
  // 3) frame-causal flash attention
  attn_fwd<<<dim3(55, NHEAD), 128, 0, stream>>>(qw, kw, vw, att);
  // 4) out = att @ Wproj + bproj   (3520x1024 @ 1024x1024)
  gemm_bf16x3<<<dim3(55, 16), 128, 0, stream>>>(att, Wproj, bproj, out,
                                                T_LEN, NHEAD * HDIM, NHEAD * HDIM);
}
